// FFTAttention_63221918597768
// MI455X (gfx1250) — matrix operand order repaired and checked by reading, not compiled, not run on hardware
//
#include <hip/hip_runtime.h>

// ---------------- problem constants ----------------
constexpr int kB   = 8;
constexpr int kC   = 64;       // channels
constexpr int kW   = 256;      // width == height
constexpr int kHW  = kW * kW;  // 65536 pixels per channel
constexpr int kQKVRows = 192;  // 64 (q) + 128 (kv) fused output channels

// ---------------- vector types ----------------
typedef _Float16 v16h __attribute__((ext_vector_type(16)));
typedef _Float16 v8h  __attribute__((ext_vector_type(8)));
typedef float    v8f  __attribute__((ext_vector_type(8)));
typedef unsigned int u32x4 __attribute__((ext_vector_type(4)));
typedef int          i32x4 __attribute__((ext_vector_type(4)));
typedef int          i32x8 __attribute__((ext_vector_type(8)));

#if defined(__has_builtin)
#if __has_builtin(__builtin_amdgcn_tensor_load_to_lds)
#define USE_TDM 1
#endif
#endif
#ifndef USE_TDM
#define USE_TDM 0
#endif

__device__ inline v16h pack16(v8h a, v8h b) {
  v16h r;
#pragma unroll
  for (int i = 0; i < 8; ++i) { r[i] = a[i]; r[i + 8] = b[i]; }
  return r;
}

// A fragment (16x32 f16): lanes 0-15 row M=lane, halves0..7->K+0..7,
// halves8..15->K+16..23; lanes 16-31 same rows with K offset +8.
__device__ inline v16h ldA(const _Float16* row, int kb) {
  const v8h* p0 = (const v8h*)(row + kb);
  const v8h* p1 = (const v8h*)(row + kb + 16);
  return pack16(*p0, *p1);
}
// B fragment (32x16 f16): lane = column N; same K layout as A fragment.
__device__ inline v16h ldB(const _Float16* row, int kb) {
  const v8h* p0 = (const v8h*)(row + kb);
  const v8h* p1 = (const v8h*)(row + kb + 16);
  return pack16(*p0, *p1);
}

// =====================================================================
// Stage 1: fused 1x1 convs  q0 = q_w @ x ; kv0 = kv_w @ x
// GEMM [192x64] x [64xHW]; one block = 64 pixels. x tile fetched by TDM.
// =====================================================================
__global__ void __launch_bounds__(256) qkv1x1_kernel(
    const float* __restrict__ x, const float* __restrict__ qw,
    const float* __restrict__ kvw, float* __restrict__ q0,
    float* __restrict__ kv0) {
  __shared__ __align__(16) _Float16 Wlds[kQKVRows * 64];  // 24 KB
  __shared__ __align__(16) _Float16 Xlds[64 * 64];        // 8 KB, [px][ch] f16
#if USE_TDM
  __shared__ __align__(16) float Xstage[64 * 64];         // 16 KB, [ch][px] f32
#endif
  const int t = threadIdx.x;
  const int wave = __builtin_amdgcn_readfirstlane(t >> 5);  // scalar wave id
  const int lane = t & 31;
  const int lhalf = lane >> 4, l15 = lane & 15;
  const int pxBase = blockIdx.x * 64;

#if USE_TDM
  // ---- Tensor Data Mover: each wave DMAs 8 channel rows of the tile ----
  {
    const unsigned ldsOff =
        (unsigned)(uintptr_t)(&Xstage[0]) + (unsigned)(wave * 8 * 64 * 4);
    const unsigned long long ga =
        (unsigned long long)(uintptr_t)(x + (size_t)wave * 8 * kHW + pxBase);
    u32x4 g0;
    g0[0] = 1u;                                     // count=1, user descriptor
    g0[1] = ldsOff;                                 // lds_addr (bytes)
    g0[2] = (unsigned)(ga & 0xffffffffull);         // global_addr[31:0]
    g0[3] = (unsigned)((ga >> 32) & 0x01ffffffull)  // global_addr[56:32]
            | (2u << 30);                           // type = 2 ("image")
    i32x8 g1;
    g1[0] = (int)(2u << 16);     // workgroup_mask=0, data_size=2 (4 bytes)
    g1[1] = 0;                   // barrier addr / tensor_dim0 low bits (0)
    g1[2] = (int)0x00400001u;    // tensor_dim0=65536 (hi16), tensor_dim1=64
    g1[3] = (int)(64u << 16);    // tile_dim0 = 64
    g1[4] = 8;                   // tile_dim1 = 8 rows per wave; tile_dim2=0
    g1[5] = (int)0x00010000u;    // tensor_dim0_stride = 65536
    g1[6] = 0;
    g1[7] = 0;
    i32x4 gz = {0, 0, 0, 0};     // groups 2/3 unused (2D tile)
#if __clang_major__ >= 23
    i32x8 gz8 = {0, 0, 0, 0, 0, 0, 0, 0};
    __builtin_amdgcn_tensor_load_to_lds(g0, g1, gz, gz, gz8, 0);
#else
    __builtin_amdgcn_tensor_load_to_lds(g0, g1, gz, gz, 0);
#endif
    __builtin_amdgcn_s_wait_tensorcnt((short)0);
  }
#endif

  // stack q_w (rows 0..63) and kv_w (rows 64..191), convert to f16
  for (int idx = t; idx < kQKVRows * 64; idx += 256) {
    const int oc = idx >> 6, ic = idx & 63;
    const float wv = (oc < 64) ? qw[oc * 64 + ic] : kvw[(oc - 64) * 64 + ic];
    Wlds[idx] = (_Float16)wv;
  }

#if USE_TDM
  __syncthreads();  // TDM slices from all waves visible
  for (int idx = t; idx < 64 * 64; idx += 256) {
    const int c = idx >> 6, p = idx & 63;
    Xlds[p * 64 + c] = (_Float16)Xstage[idx];  // transpose + f16
  }
#else
  {
    const int c = t >> 2;
    const int pc = (t & 3) * 16;
    const float* src = x + (size_t)c * kHW + pxBase + pc;
    __builtin_prefetch(src, 0, 0);
#pragma unroll
    for (int u = 0; u < 16; ++u) Xlds[(pc + u) * 64 + c] = (_Float16)src[u];
  }
#endif
  __syncthreads();

  // 12 oc-tiles x 4 px-tiles = 48 tiles over 8 waves, fully unrolled
#pragma unroll
  for (int tt = 0; tt < 6; ++tt) {
    const int tile = wave + tt * 8;       // scalar
    const int ot = tile >> 2, pt = tile & 3;
    const _Float16* arow = &Wlds[(ot * 16 + l15) * 64];
    const _Float16* brow = &Xlds[(pt * 16 + l15) * 64];
    v8f acc = {0.f, 0.f, 0.f, 0.f, 0.f, 0.f, 0.f, 0.f};
#pragma unroll
    for (int k0 = 0; k0 < 64; k0 += 32) {
      v16h A = ldA(arow, k0 + (lhalf << 3));
      v16h Bf = ldB(brow, k0 + (lhalf << 3));
      acc = __builtin_amdgcn_wmma_f32_16x16x32_f16(
          false, A, false, Bf, (short)0, acc, false, false);
    }
    // scalar select of destination; straight-line stores
    float* dst;
    int ocl;
    if (ot < 4) { dst = q0;  ocl = ot * 16; }
    else        { dst = kv0; ocl = ot * 16 - 64; }
    const int px = pxBase + pt * 16 + l15;
    const int ocb = ocl + (lhalf << 3);
#pragma unroll
    for (int r = 0; r < 8; ++r)
      dst[(size_t)(ocb + r) * kHW + px] = acc[r];
  }
}

// =====================================================================
// Stage 2: depthwise 3x3 on q0 (pad 1). 256 blocks per channel.
// =====================================================================
__global__ void __launch_bounds__(256) dw3x3_kernel(
    const float* __restrict__ in, const float* __restrict__ w,
    float* __restrict__ out) {
  __shared__ float sw[9];
  const int c = blockIdx.x >> 8;
  const int p = ((blockIdx.x & 255) << 8) + threadIdx.x;
  if (threadIdx.x < 9) sw[threadIdx.x] = w[c * 9 + threadIdx.x];
  __syncthreads();
  const int y = p >> 8, xx0 = p & 255;
  const float* base = in + (size_t)c * kHW;
  float acc = 0.f;
#pragma unroll
  for (int dy = -1; dy <= 1; ++dy) {
    const int yy = y + dy;
    if ((unsigned)yy < (unsigned)kW) {
      const float* row = base + yy * kW;
#pragma unroll
      for (int dx = -1; dx <= 1; ++dx) {
        const int xx = xx0 + dx;
        if ((unsigned)xx < (unsigned)kW)
          acc += sw[(dy + 1) * 3 + (dx + 1)] * row[xx];
      }
    }
  }
  out[(size_t)c * kHW + p] = acc;
}

// =====================================================================
// Stage 3: depthwise 7x7 on kv0 (pad 3), split into k (c<64) and v.
// =====================================================================
__global__ void __launch_bounds__(256) dw7x7_kernel(
    const float* __restrict__ in, const float* __restrict__ w,
    float* __restrict__ kout, float* __restrict__ vout) {
  __shared__ float sw[49];
  const int c = blockIdx.x >> 8;  // 0..127
  const int p = ((blockIdx.x & 255) << 8) + threadIdx.x;
  if (threadIdx.x < 49) sw[threadIdx.x] = w[c * 49 + threadIdx.x];
  __syncthreads();
  const int y = p >> 8, xx0 = p & 255;
  const float* base = in + (size_t)c * kHW;
  __builtin_prefetch(base + p + 4 * kW, 0, 0);  // global_prefetch_b8
  float acc = 0.f;
#pragma unroll
  for (int dy = -3; dy <= 3; ++dy) {
    const int yy = y + dy;
    if ((unsigned)yy < (unsigned)kW) {
      const float* row = base + yy * kW;
#pragma unroll
      for (int dx = -3; dx <= 3; ++dx) {
        const int xx = xx0 + dx;
        if ((unsigned)xx < (unsigned)kW)
          acc += sw[(dy + 3) * 7 + (dx + 3)] * row[xx];
      }
    }
  }
  if (c < 64) kout[(size_t)c * kHW + p] = acc;
  else        vout[(size_t)(c - 64) * kHW + p] = acc;
}

// =====================================================================
// Stage 4: per-patch 8x8 circular convolution (== ifft2(fft2(q)*fft2(k)).real)
// =====================================================================
__global__ void __launch_bounds__(256) circconv_kernel(
    const float* __restrict__ q, const float* __restrict__ k,
    float* __restrict__ a) {
  __shared__ float qs[4][64];
  __shared__ float ks[4][64];
  const int t = threadIdx.x;
  const int sp = t >> 6;
  const int r = t & 63;
  const int g = blockIdx.x * 4 + sp;  // (channel, patch) id
  const int c = g >> 10;
  const int pidx = g & 1023;
  const int py = pidx >> 5, px = pidx & 31;
  const size_t base = (size_t)c * kHW + (size_t)(py * 8) * kW + px * 8;
  const int i = r >> 3, j = r & 7;
  qs[sp][r] = q[base + i * kW + j];
  ks[sp][r] = k[base + i * kW + j];
  __syncthreads();
  float acc = 0.f;
#pragma unroll
  for (int ii = 0; ii < 8; ++ii) {
    const float* qr = &qs[sp][ii * 8];
    const float* kr = &ks[sp][((i - ii) & 7) * 8];
#pragma unroll
    for (int jj = 0; jj < 8; ++jj) acc += qr[jj] * kr[(j - jj) & 7];
  }
  a[base + i * kW + j] = acc;
}

// =====================================================================
// Stage 5: fused LayerNorm(channels) -> * v -> 1x1 proj (WMMA) -> + bias
// =====================================================================
__global__ void __launch_bounds__(256) lnproj_kernel(
    const float* __restrict__ a, const float* __restrict__ v,
    const float* __restrict__ lnw, const float* __restrict__ lnb,
    const float* __restrict__ pw, const float* __restrict__ pb,
    float* __restrict__ out) {
  __shared__ __align__(16) float aT[64 * 64];     // [px][ch]
  __shared__ __align__(16) float vT[64 * 64];     // [px][ch]
  __shared__ __align__(16) _Float16 bT[64 * 64];  // [px][ch]
  __shared__ __align__(16) _Float16 Wp[64 * 64];  // [oc][ic]
  __shared__ float mu[64], rs[64];
  const int t = threadIdx.x;
  const int wave = __builtin_amdgcn_readfirstlane(t >> 5);
  const int lane = t & 31;
  const int lhalf = lane >> 4, l15 = lane & 15;
  const int pxBase = blockIdx.x * 64;
  {
    const int c = t >> 2, pc = (t & 3) * 16;
    const float* sa = a + (size_t)c * kHW + pxBase + pc;
    const float* sv = v + (size_t)c * kHW + pxBase + pc;
#pragma unroll
    for (int u = 0; u < 16; ++u) {
      aT[(pc + u) * 64 + c] = sa[u];
      vT[(pc + u) * 64 + c] = sv[u];
    }
  }
  for (int idx = t; idx < 64 * 64; idx += 256) Wp[idx] = (_Float16)pw[idx];
  __syncthreads();
  if (t < 64) {  // per-pixel mean/var over 64 channels
    const float* row = &aT[t * 64];
    float s = 0.f;
#pragma unroll 8
    for (int c = 0; c < 64; ++c) s += row[c];
    const float m = s * (1.f / 64.f);
    float v2 = 0.f;
#pragma unroll 8
    for (int c = 0; c < 64; ++c) { const float d = row[c] - m; v2 += d * d; }
    mu[t] = m;
    rs[t] = rsqrtf(v2 * (1.f / 64.f) + 1e-5f);
  }
  __syncthreads();
  {
    const int c = t >> 2, pc = (t & 3) * 16;
    const float wl = lnw[c], bl = lnb[c];
#pragma unroll
    for (int u = 0; u < 16; ++u) {
      const int p = pc + u;
      const float val = (aT[p * 64 + c] - mu[p]) * rs[p] * wl + bl;
      bT[p * 64 + c] = (_Float16)(val * vT[p * 64 + c]);
    }
  }
  __syncthreads();
#pragma unroll
  for (int tt = 0; tt < 2; ++tt) {
    const int tile = wave + tt * 8;  // scalar: 4 oc-tiles x 4 px-tiles
    const int ot = tile >> 2, pt = tile & 3;
    const _Float16* arow = &Wp[(ot * 16 + l15) * 64];
    const _Float16* brow = &bT[(pt * 16 + l15) * 64];
    v8f acc = {0.f, 0.f, 0.f, 0.f, 0.f, 0.f, 0.f, 0.f};
#pragma unroll
    for (int k0 = 0; k0 < 64; k0 += 32) {
      v16h A = ldA(arow, k0 + (lhalf << 3));
      v16h Bf = ldB(brow, k0 + (lhalf << 3));
      acc = __builtin_amdgcn_wmma_f32_16x16x32_f16(
          false, A, false, Bf, (short)0, acc, false, false);
    }
    const int px = pxBase + pt * 16 + l15;
    const int ocb = ot * 16 + (lhalf << 3);
#pragma unroll
    for (int r = 0; r < 8; ++r) {
      const int oc = ocb + r;
      out[(size_t)oc * kHW + px] = acc[r] + pb[oc];
    }
  }
}

// =====================================================================
// Host launcher: 5 stages per batch, workspace buffers reused per batch.
// =====================================================================
extern "C" void kernel_launch(void* const* d_in, const int* in_sizes, int n_in,
                              void* d_out, int out_size, void* d_ws,
                              size_t ws_size, hipStream_t stream) {
  (void)in_sizes; (void)n_in; (void)out_size; (void)ws_size;
  const float* x    = (const float*)d_in[0];
  const float* qw   = (const float*)d_in[1];
  const float* qdw  = (const float*)d_in[2];
  const float* kvw  = (const float*)d_in[3];
  const float* kvdw = (const float*)d_in[4];
  const float* lnw  = (const float*)d_in[5];
  const float* lnb  = (const float*)d_in[6];
  const float* pw   = (const float*)d_in[7];
  const float* pb   = (const float*)d_in[8];
  float* out = (float*)d_out;
  float* ws  = (float*)d_ws;

  const size_t CH = (size_t)kC * kHW;
  float* q0  = ws;
  float* kv0 = ws + CH;
  float* q   = ws + 3 * CH;
  float* k   = ws + 4 * CH;
  float* v   = ws + 5 * CH;
  float* a   = ws + 6 * CH;  // total ws use: 7*CH floats = 117 MB

  for (int b = 0; b < kB; ++b) {
    const float* xb = x + (size_t)b * CH;
    float* ob = out + (size_t)b * CH;
    qkv1x1_kernel<<<dim3(kHW / 64), dim3(256), 0, stream>>>(xb, qw, kvw, q0, kv0);
    dw3x3_kernel<<<dim3(kC * (kHW / 256)), dim3(256), 0, stream>>>(q0, qdw, q);
    dw7x7_kernel<<<dim3(2 * kC * (kHW / 256)), dim3(256), 0, stream>>>(kv0, kvdw, k, v);
    circconv_kernel<<<dim3(kC * 1024 / 4), dim3(256), 0, stream>>>(q, k, a);
    lnproj_kernel<<<dim3(kHW / 64), dim3(256), 0, stream>>>(a, v, lnw, lnb, pw, pb, ob);
  }
}
